// RelativePositionCrossAttention_28003186770302
// MI455X (gfx1250) — matrix operand order repaired and checked by reading, not compiled, not run on hardware
//
#include <hip/hip_runtime.h>
#include <hip/hip_bf16.h>

// ---------------------------------------------------------------------------
// RelativePositionCrossAttention for MI455X (gfx1250, wave32, WMMA f16)
//   B=4, qL=1024 (32x32), kvL=4096 (64x64), C=256, H=8, hd=32
// Pipeline:
//   1) proj_kernel<1>: Qh  = f16[(B,H,qL,hd)]  = (q @ Wq^T + bq) * hd^-0.5
//   2) proj_kernel<2>: Kh/Vh f16[(B,H,kvL,hd)] = kv @ Wkv^T + bkv (split)
//   3) attn_kernel   : flash-style fused attention + rel-pos bias -> Ctx f32
//        - async global->LDS double-buffered K staging (ASYNCcnt) if available
//        - DPP16 row reductions for online softmax (no ds_bpermute)
//   4) proj_kernel<0>: out = Ctx @ Wp^T + bp   (f32)
// ---------------------------------------------------------------------------

typedef _Float16 f16_t;
typedef __attribute__((ext_vector_type(16))) _Float16 v16h;
typedef __attribute__((ext_vector_type(8)))  _Float16 h8;
typedef __attribute__((ext_vector_type(8)))  float    v8f;

#define DIMC      256
#define NUM_HEADS 8
#define HEAD_DIM  32
#define QL        1024
#define KVL       4096
#define NCHUNK    (KVL / 32)

#if __has_builtin(__builtin_amdgcn_global_load_async_to_lds_b128) && \
    __has_builtin(__builtin_amdgcn_s_wait_asynccnt)
#define USE_ASYNC_LDS 1
typedef int v4i __attribute__((vector_size(16)));              // builtin param type
typedef __attribute__((address_space(3))) v4i* lds_v4i_p;      // __shared__ v4i*
#else
#define USE_ASYNC_LDS 0
#endif

#define WMMA_F16(a, b, c) \
  __builtin_amdgcn_wmma_f32_16x16x32_f16(false, (a), false, (b), (short)0, (c), false, false)

// --- fragment pack helpers --------------------------------------------------
// A-matrix f16 16x32:           lane m = lane&15; k = e + (lane>=16?16:0)
// B-matrix f16 32x16:           lane n = lane&15; k = e + (lane>=16?16:0)

static __device__ inline v16h pack16_f32(const float* __restrict__ p) {
  v16h r;
#pragma unroll
  for (int i = 0; i < 16; ++i) r[i] = (_Float16)p[i];
  return r;
}
static __device__ inline v16h pack8x2_f32(const float* __restrict__ p0,
                                          const float* __restrict__ p1) {
  v16h r;
#pragma unroll
  for (int i = 0; i < 8; ++i) { r[i] = (_Float16)p0[i]; r[8 + i] = (_Float16)p1[i]; }
  return r;
}
static __device__ inline v16h ldh16(const f16_t* p) {  // 16 consecutive f16 (32B)
  h8 a = *(const h8*)p;
  h8 b = *(const h8*)(p + 8);
  v16h r;
#pragma unroll
  for (int i = 0; i < 8; ++i) { r[i] = a[i]; r[8 + i] = b[i]; }
  return r;
}
static __device__ inline v16h ldh8x2(const f16_t* p0, const f16_t* p1) {
  h8 a = *(const h8*)p0;
  h8 b = *(const h8*)p1;
  v16h r;
#pragma unroll
  for (int i = 0; i < 8; ++i) { r[i] = a[i]; r[8 + i] = b[i]; }
  return r;
}

// --- DPP16 reductions over each 16-lane row (rows reduce independently) -----
// quad_perm xor1 (0xB1), quad_perm xor2 (0x4E), row_ror:4 (0x124), row_ror:8 (0x128)
static __device__ inline float row16_max(float x) {
  int t;
  t = __builtin_amdgcn_update_dpp(0, __float_as_int(x), 0xB1,  0xF, 0xF, true);
  x = fmaxf(x, __int_as_float(t));
  t = __builtin_amdgcn_update_dpp(0, __float_as_int(x), 0x4E,  0xF, 0xF, true);
  x = fmaxf(x, __int_as_float(t));
  t = __builtin_amdgcn_update_dpp(0, __float_as_int(x), 0x124, 0xF, 0xF, true);
  x = fmaxf(x, __int_as_float(t));
  t = __builtin_amdgcn_update_dpp(0, __float_as_int(x), 0x128, 0xF, 0xF, true);
  return fmaxf(x, __int_as_float(t));
}
static __device__ inline float row16_sum(float x) {
  int t;
  t = __builtin_amdgcn_update_dpp(0, __float_as_int(x), 0xB1,  0xF, 0xF, true);
  x += __int_as_float(t);
  t = __builtin_amdgcn_update_dpp(0, __float_as_int(x), 0x4E,  0xF, 0xF, true);
  x += __int_as_float(t);
  t = __builtin_amdgcn_update_dpp(0, __float_as_int(x), 0x124, 0xF, 0xF, true);
  x += __int_as_float(t);
  t = __builtin_amdgcn_update_dpp(0, __float_as_int(x), 0x128, 0xF, 0xF, true);
  return x + __int_as_float(t);
}

// ---------------------------------------------------------------------------
// Generic projection GEMM: out[M,N] = A[M,C] @ W[N,C]^T + bias[N]
// One wave computes a 16(M) x 64(N) tile, k-steps of 32, 4 WMMAs per step.
// ---------------------------------------------------------------------------
template <int MODE>
__global__ void __launch_bounds__(32)
proj_kernel(const float* __restrict__ A, const float* __restrict__ W,
            const float* __restrict__ bias, int Cdim, float scale,
            f16_t* __restrict__ out0, f16_t* __restrict__ out1,
            float* __restrict__ outf, int Nld)
{
  const int lane  = threadIdx.x & 31;
  const int m0    = blockIdx.x * 16;
  const int n0    = blockIdx.y * 64;
  const int mrow  = lane & 15;
  const int abase = (lane >> 4) << 3;   // unused; A-frag uses bbase (0 / 16)
  const int bbase = (lane >> 4) << 4;   // B-frag k offset (0 / 16)
  const int ncol  = lane & 15;
  const int mhalf = (lane >> 4) << 3;   // C-frag row offset (0 / 8)

  v8f acc[4] = {};
  const float* arow = A + (size_t)(m0 + mrow) * Cdim;
  const float* w0   = W + (size_t)(n0 + ncol) * Cdim;

  for (int k0 = 0; k0 < Cdim; k0 += 32) {
    v16h a = pack16_f32(arow + k0 + bbase);
#pragma unroll
    for (int s = 0; s < 4; ++s) {
      v16h bf = pack16_f32(w0 + (size_t)s * 16 * Cdim + k0 + bbase);
      acc[s] = WMMA_F16(a, bf, acc[s]);
    }
  }

#pragma unroll
  for (int s = 0; s < 4; ++s) {
    const int n  = n0 + s * 16 + ncol;
    const float bn = bias[n];
#pragma unroll
    for (int r = 0; r < 8; ++r) {
      const int m = m0 + r + mhalf;
      float v = acc[s][r] + bn;
      if (MODE == 0) {
        outf[(size_t)m * Nld + n] = v;
      } else if (MODE == 1) {
        v *= scale;
        const int h = n >> 5, d = n & 31;
        const int bb = m >> 10, q = m & (QL - 1);
        out0[(((size_t)(bb * NUM_HEADS + h) * QL) + q) * HEAD_DIM + d] = (f16_t)v;
      } else {
        const int sidx = n >> 8;                 // 0 -> K, 1 -> V
        const int h = (n >> 5) & 7, d = n & 31;
        const int bb = m >> 12, kk = m & (KVL - 1);
        f16_t* dst = sidx ? out1 : out0;
        dst[(((size_t)(bb * NUM_HEADS + h) * KVL) + kk) * HEAD_DIM + d] = (f16_t)v;
      }
    }
  }
}

// ---------------------------------------------------------------------------
// Fused attention with relative-position bias (flash-style, online softmax).
// Block = 128 threads = 4 waves; block handles (b, h, 64 q rows); each wave
// owns a 16-row q tile. kv streamed in 32-wide chunks, double-buffered:
// chunk i+1 is fetched (K: async global->LDS, V: to registers) while chunk i
// is computed.
// ---------------------------------------------------------------------------
__global__ void __launch_bounds__(128)
attn_kernel(const f16_t* __restrict__ Qh, const f16_t* __restrict__ Kh,
            const f16_t* __restrict__ Vh, const float* __restrict__ bias_table,
            const int* __restrict__ rel_index, float* __restrict__ Ctx)
{
  __shared__ f16_t Kbuf[2][32 * 32];   // [kv_local][d]  row-major
  __shared__ f16_t Vt[2][32 * 32];     // [d][kv_local]  transposed
  __shared__ f16_t Pbuf[4][16 * 32];   // per-wave P tile [m][kv_local]

  const int tid  = threadIdx.x;
  const int lane = tid & 31;
  const int wv   = tid >> 5;
  const int bid  = blockIdx.x;
  const int q64  = bid & 15;
  const int h    = (bid >> 4) & 7;
  const int b    = bid >> 7;
  const int q0   = q64 * 64 + wv * 16;

  const int mrow  = lane & 15;
  const int abase = (lane >> 4) << 3;
  const int bbase = (lane >> 4) << 4;
  const int ncol  = lane & 15;
  const int mhalf = (lane >> 4) << 3;

  const size_t bh = (size_t)(b * NUM_HEADS + h);

  // Q fragment for this wave's 16 rows (persistent across the kv loop)
  const f16_t* qp = Qh + (bh * QL + q0 + mrow) * HEAD_DIM;
  const v16h aq = ldh16(qp + bbase);

  v8f o0 = {}, o1 = {};
  float rmax[8], rsum[8];
#pragma unroll
  for (int r = 0; r < 8; ++r) { rmax[r] = -3.0e38f; rsum[r] = 0.f; }

  const int* relp = rel_index + (size_t)q0 * KVL;

  // staging: each thread owns one 8-f16 (16B) segment of the 32x32 chunk
  const int skv = tid >> 2;         // kv row 0..31
  const int sd  = (tid & 3) << 3;   // d offset 0/8/16/24
  const f16_t* ksrc = Kh + (bh * KVL + skv) * HEAD_DIM + sd;
  const f16_t* vsrc = Vh + (bh * KVL + skv) * HEAD_DIM + sd;

  // ---- pipeline prologue: fetch chunk 0 ----
  h8 vreg = *(const h8*)(vsrc);
#if USE_ASYNC_LDS
  __builtin_amdgcn_global_load_async_to_lds_b128(
      (v4i*)ksrc, (lds_v4i_p)&Kbuf[0][skv * 32 + sd], 0, 0);
#else
  h8 kreg = *(const h8*)(ksrc);
#endif

  for (int i = 0; i < NCHUNK; ++i) {
    const int kc  = i * 32;
    const int buf = i & 1;

    // --- commit staged V (transpose) [and K in the sync fallback] ---
#pragma unroll
    for (int j = 0; j < 8; ++j) Vt[buf][(sd + j) * 32 + skv] = vreg[j];
#if !USE_ASYNC_LDS
    *(h8*)(&Kbuf[buf][skv * 32 + sd]) = kreg;
#endif

    // --- issue fetch of chunk i+1 while chunk i computes ---
    if (i + 1 < NCHUNK) {
      vreg = *(const h8*)(vsrc + (size_t)(kc + 32) * HEAD_DIM);
#if USE_ASYNC_LDS
      __builtin_amdgcn_global_load_async_to_lds_b128(
          (v4i*)(ksrc + (size_t)(kc + 32) * HEAD_DIM),
          (lds_v4i_p)&Kbuf[buf ^ 1][skv * 32 + sd], 0, 0);
#else
      kreg = *(const h8*)(ksrc + (size_t)(kc + 32) * HEAD_DIM);
#endif
      if (kc + 64 < KVL) {  // keep the L2 stream warm two chunks ahead
        __builtin_prefetch(ksrc + (size_t)(kc + 64) * HEAD_DIM, 0, 1);
        __builtin_prefetch(vsrc + (size_t)(kc + 64) * HEAD_DIM, 0, 1);
      }
    }
#if USE_ASYNC_LDS
    if (i + 1 < NCHUNK) __builtin_amdgcn_s_wait_asynccnt(1);  // chunk i done
    else                __builtin_amdgcn_s_wait_asynccnt(0);
#endif
    __syncthreads();

    // --- S = Q K^T for two 16x16 tiles of this 32-wide kv chunk ---
    const v8f zero = {};
    v16h bk0 = ldh16(&Kbuf[buf][ncol * 32 + bbase]);
    v16h bk1 = ldh16(&Kbuf[buf][(16 + ncol) * 32 + bbase]);
    v8f s0 = WMMA_F16(aq, bk0, zero);
    v8f s1 = WMMA_F16(aq, bk1, zero);

    // --- bias add + online softmax (DPP16 row reductions) ---
    f16_t* pb = Pbuf[wv];
#pragma unroll
    for (int r = 0; r < 8; ++r) {
      const int mm = r + mhalf;
      const int i0 = relp[(size_t)mm * KVL + kc + ncol];
      const int i1 = relp[(size_t)mm * KVL + kc + 16 + ncol];
      const float v0 = s0[r] + bias_table[i0 * NUM_HEADS + h];
      const float v1 = s1[r] + bias_table[i1 * NUM_HEADS + h];
      const float mx = row16_max(fmaxf(v0, v1));
      const float nm = fmaxf(rmax[r], mx);
      const float cf = __expf(rmax[r] - nm);
      rmax[r] = nm;
      const float p0 = __expf(v0 - nm);
      const float p1 = __expf(v1 - nm);
      rsum[r] = rsum[r] * cf + row16_sum(p0 + p1);
      o0[r] *= cf;
      o1[r] *= cf;
      pb[mm * 32 + ncol]      = (f16_t)p0;   // C-layout -> LDS
      pb[mm * 32 + 16 + ncol] = (f16_t)p1;
    }

    // --- O += P @ V (P re-read from LDS in A-fragment layout) ---
    v16h ap  = ldh16(pb + mrow * 32 + bbase);
    v16h bv0 = ldh16(&Vt[buf][ncol * 32 + bbase]);
    v16h bv1 = ldh16(&Vt[buf][(16 + ncol) * 32 + bbase]);
    o0 = WMMA_F16(ap, bv0, o0);
    o1 = WMMA_F16(ap, bv1, o1);

    __syncthreads();
  }

  // --- normalize + write Ctx[b, q, h*32 + d] ---
  float* cbase = Ctx + ((size_t)b * QL) * DIMC + h * HEAD_DIM;
#pragma unroll
  for (int r = 0; r < 8; ++r) {
    const int mm = r + mhalf;
    const float inv = 1.0f / rsum[r];
    float* cp = cbase + (size_t)(q0 + mm) * DIMC;
    cp[ncol]      = o0[r] * inv;
    cp[16 + ncol] = o1[r] * inv;
  }
}

// ---------------------------------------------------------------------------
extern "C" void kernel_launch(void* const* d_in, const int* in_sizes, int n_in,
                              void* d_out, int out_size, void* d_ws, size_t ws_size,
                              hipStream_t stream)
{
  (void)in_sizes; (void)n_in; (void)out_size; (void)ws_size;
  const float* q   = (const float*)d_in[0];   // [4,1024,256]
  const float* kv  = (const float*)d_in[1];   // [4,4096,256]
  const float* Wq  = (const float*)d_in[2];   // [256,256]
  const float* bq  = (const float*)d_in[3];   // [256]
  const float* Wkv = (const float*)d_in[4];   // [512,256]
  const float* bkv = (const float*)d_in[5];   // [512]
  const float* bt  = (const float*)d_in[6];   // [16129,8]
  const float* Wp  = (const float*)d_in[7];   // [256,256]
  const float* bp  = (const float*)d_in[8];   // [256]
  const int*   rel = (const int*)d_in[9];     // [1024,4096]
  float* out = (float*)d_out;                 // [4,1024,256] f32

  char* ws = (char*)d_ws;
  f16_t* Qh  = (f16_t*)ws;                        //  2 MiB  [B,H,qL,hd]
  f16_t* Kh  = (f16_t*)(ws + (2ull  << 20));      //  8 MiB  [B,H,kvL,hd]
  f16_t* Vh  = (f16_t*)(ws + (10ull << 20));      //  8 MiB  [B,H,kvL,hd]
  float* Ctx = (float*)(ws + (18ull << 20));      //  4 MiB  [B,qL,C]

  const float scale = 0.17677669529663687f;       // 32^-0.5

  // Q projection: M=4096, N=256
  proj_kernel<1><<<dim3(256, 4), 32, 0, stream>>>(q, Wq, bq, DIMC, scale,
                                                  Qh, nullptr, nullptr, DIMC);
  // KV projection: M=16384, N=512
  proj_kernel<2><<<dim3(1024, 8), 32, 0, stream>>>(kv, Wkv, bkv, DIMC, 1.f,
                                                   Kh, Vh, nullptr, 2 * DIMC);
  // Fused attention: B*H*(qL/64) = 512 blocks of 4 waves
  attn_kernel<<<512, 128, 0, stream>>>(Qh, Kh, Vh, bt, rel, Ctx);
  // Output projection: M=4096, N=256
  proj_kernel<0><<<dim3(256, 4), 32, 0, stream>>>(Ctx, Wp, bp, DIMC, 1.f,
                                                  nullptr, nullptr, out, DIMC);
}
